// SelfAttention_66606352827323
// MI455X (gfx1250) — matrix operand order repaired and checked by reading, not compiled, not run on hardware
//
#include <hip/hip_runtime.h>
#include <hip/hip_bf16.h>

// ---------------------------------------------------------------------------
// Causal multi-head self-attention, bf16 WMMA pipeline for gfx1250 (MI455X).
// N=2, S=2048, E=1024, H=16, D=64.
//  - All GEMMs + attention use v_wmma_f32_16x16x32_bf16 (fp32 accumulate).
//  - Attention K/V tiles staged via global_load_async_to_lds_b128 (ASYNCcnt),
//    double buffered and shared by the 4 waves of a block.
// ---------------------------------------------------------------------------

typedef __bf16 bf16_t;
typedef __attribute__((ext_vector_type(16))) __bf16 v16bf;
typedef __attribute__((ext_vector_type(8)))  __bf16 v8bf;
typedef __attribute__((ext_vector_type(8)))  float  v8f;

union ABFrag { v16bf v; v8bf h[2]; };

__device__ __forceinline__ v8f vzero8() {
  v8f z = {0.f,0.f,0.f,0.f,0.f,0.f,0.f,0.f};
  return z;
}

// A-matrix fragment (16x32 bf16, MxK): lane(0-15) holds row M=lane,
// K = {0..7, 16..23}; lane(16-31) row M=lane-16, K = {8..15, 24..31}.
__device__ __forceinline__ v16bf load_a_frag(const bf16_t* base, int ld) {
  const int lane = threadIdx.x & 31;
  const int half = lane >> 4, r = lane & 15;
  const bf16_t* p = base + r * ld + half * 8;
  ABFrag f;
  f.h[0] = *(const v8bf*)(p);
  f.h[1] = *(const v8bf*)(p + 16);
  return f.v;
}

// B-matrix fragment (32x16 bf16, KxN) from Bt[n][k] row-major (K contiguous):
// lane(0-15) col N=lane, K={0..7,16..23}; lane(16-31) col N=lane-16, K={8..15,24..31}.
__device__ __forceinline__ v16bf load_b_frag(const bf16_t* base, int ld) {
  const int lane = threadIdx.x & 31;
  const int half = lane >> 4, n = lane & 15;
  const bf16_t* p = base + n * ld + half * 8;
  ABFrag f;
  f.h[0] = *(const v8bf*)(p);
  f.h[1] = *(const v8bf*)(p + 16);
  return f.v;
}

__device__ __forceinline__ v8f wmma_bf16(v16bf a, v16bf b, v8f c) {
  return __builtin_amdgcn_wmma_f32_16x16x32_bf16(
      false, a, false, b, (short)0, c, false, false);
}

// Async DMA of one 16B chunk: global -> LDS, tracked with ASYNCcnt.
__device__ __forceinline__ void async_b128(uint32_t lds_byte_off, const void* gaddr) {
  asm volatile("global_load_async_to_lds_b128 %0, %1, off"
               :: "v"(lds_byte_off), "v"(gaddr) : "memory");
}

// ---------------------------------------------------------------------------
// fp32 -> bf16 conversion
// ---------------------------------------------------------------------------
__global__ void __launch_bounds__(256) cvt_f32_bf16(const float* __restrict__ in,
                                                    bf16_t* __restrict__ out, int n) {
  int i = blockIdx.x * 256 + threadIdx.x;
  if (i < n) out[i] = (bf16_t)in[i];
}

// ---------------------------------------------------------------------------
// GEMM:  C[M,Nn] = A[M,K](bf16) x W[Nn,K]^T(bf16) + bias   (x @ W.T + b)
// 4 waves/block, 64x64 tile per wave, 16 WMMAs per K=32 step.
// Epilogue modes: 0 fp32 [M,Nn]; 1 bf16 [N,H,S,D]; 2 bf16 [N,H,D,S].
// ---------------------------------------------------------------------------
__global__ void __launch_bounds__(128) gemm_bf16_wmma(
    const bf16_t* __restrict__ A, const bf16_t* __restrict__ Wt,
    const float* __restrict__ bias, void* __restrict__ out,
    int M, int Nn, int K, int mode)
{
  const int lane = threadIdx.x & 31;
  const int wave = threadIdx.x >> 5;
  const int m0 = (blockIdx.x * 4 + wave) * 64;
  const int n0 = blockIdx.y * 64;
  if (m0 >= M) return;

  v8f acc[4][4];
#pragma unroll
  for (int i = 0; i < 4; ++i)
#pragma unroll
    for (int j = 0; j < 4; ++j) acc[i][j] = vzero8();

  for (int k0 = 0; k0 < K; k0 += 32) {
    if (k0 + 32 < K) {  // hide L2 latency for the next K step
      __builtin_prefetch(A  + (size_t)(m0 + lane) * K + k0 + 32, 0, 3);
      __builtin_prefetch(Wt + (size_t)(n0 + lane) * K + k0 + 32, 0, 3);
    }
    v16bf af[4], bf[4];
#pragma unroll
    for (int i = 0; i < 4; ++i)
      af[i] = load_a_frag(A + (size_t)(m0 + i * 16) * K + k0, K);
#pragma unroll
    for (int j = 0; j < 4; ++j)
      bf[j] = load_b_frag(Wt + (size_t)(n0 + j * 16) * K + k0, K);
#pragma unroll
    for (int i = 0; i < 4; ++i)
#pragma unroll
      for (int j = 0; j < 4; ++j)
        acc[i][j] = wmma_bf16(af[i], bf[j], acc[i][j]);
  }

  const int half = lane >> 4, nl = lane & 15;
#pragma unroll
  for (int i = 0; i < 4; ++i)
#pragma unroll
    for (int j = 0; j < 4; ++j)
#pragma unroll
      for (int e = 0; e < 8; ++e) {
        int row = m0 + i * 16 + half * 8 + e;   // token index (nb*S + s)
        int col = n0 + j * 16 + nl;             // feature index (h*64 + d)
        float v = acc[i][j][e] + bias[col];
        if (mode == 0) {
          ((float*)out)[(size_t)row * Nn + col] = v;
        } else {
          int nb = row >> 11, s = row & 2047;   // S = 2048
          int h  = col >> 6,  d = col & 63;     // D = 64
          if (mode == 1)
            ((bf16_t*)out)[(((size_t)(nb * 16 + h) * 2048) + s) * 64 + d] = (bf16_t)v;
          else
            ((bf16_t*)out)[(((size_t)(nb * 16 + h) * 64) + d) * 2048 + s] = (bf16_t)v;
        }
      }
}

// ---------------------------------------------------------------------------
// Flash attention (causal).  One block = 128 query rows of one (batch, head);
// wave w owns rows q0 = bx*128 + w*32.  All 4 waves share double-buffered
// K/V tiles staged in LDS by async DMA.
// Q,K: [N,H,S,D] bf16;  VT: [N,H,D,S] bf16;  Om: [N,S,E] bf16.
// ---------------------------------------------------------------------------
__global__ void __launch_bounds__(128) attn_flash_wmma(
    const bf16_t* __restrict__ Qm, const bf16_t* __restrict__ Km,
    const bf16_t* __restrict__ VTm, bf16_t* __restrict__ Om)
{
  constexpr int S = 2048, D = 64;
  const int tid  = threadIdx.x;
  const int lane = tid & 31;
  const int wave = tid >> 5;
  const int nh = blockIdx.y;                 // n*H + h
  const int nb = nh >> 4, h = nh & 15;
  const int q0b = blockIdx.x * 128;
  const int q0  = q0b + wave * 32;
  const int half = lane >> 4, nl = lane & 15;

  const bf16_t* Qh = Qm  + (size_t)nh * S * D;
  const bf16_t* Kh = Km  + (size_t)nh * S * D;
  const bf16_t* Vh = VTm + (size_t)nh * D * S;

  // LDS: K/V tiles (64x64 bf16, double buffered, block-shared) + per-wave P.
  __shared__ __align__(16) bf16_t KtL[2][64 * 64];   // [key][d]
  __shared__ __align__(16) bf16_t VtL[2][64 * 64];   // [d][key]
  __shared__ __align__(16) bf16_t Plds[4][32 * 64];
  bf16_t* P = Plds[wave];
  const uint32_t ldsK0 = (uint32_t)(size_t)&KtL[0][0];
  const uint32_t ldsK1 = (uint32_t)(size_t)&KtL[1][0];
  const uint32_t ldsV0 = (uint32_t)(size_t)&VtL[0][0];
  const uint32_t ldsV1 = (uint32_t)(size_t)&VtL[1][0];

  // Stage one 64-key tile pair (K: 64 rows x 128B, V^T: 64 rows x 128B)
  // cooperatively: 512 chunks / 128 threads = 4 async ops each per tile.
  auto stage = [&](int kbase, uint32_t ldsK, uint32_t ldsV) {
#pragma unroll
    for (int i = 0; i < 4; ++i) {
      const int c = tid + i * 128;             // chunk 0..511
      const int row = c >> 3, ch = c & 7;      // 64 rows x 8 x 16B
      async_b128(ldsK + (uint32_t)(row * 128 + ch * 16),
                 Kh + (size_t)(kbase + row) * D + ch * 8);
    }
#pragma unroll
    for (int i = 0; i < 4; ++i) {
      const int c = tid + i * 128;
      const int row = c >> 3, ch = c & 7;      // row = d, ch along keys
      async_b128(ldsV + (uint32_t)(row * 128 + ch * 16),
                 Vh + (size_t)row * S + kbase + ch * 8);
    }
  };

  // Q fragments held in registers: 2 M-frags x 2 head-dim chunks of 32.
  v16bf qf[2][2];
#pragma unroll
  for (int mf = 0; mf < 2; ++mf)
#pragma unroll
    for (int kc = 0; kc < 2; ++kc)
      qf[mf][kc] = load_a_frag(Qh + (size_t)(q0 + mf * 16) * D + kc * 32, D);

  v8f o[2][4];
  float mrow[2][8], lrow[2][8];
#pragma unroll
  for (int mf = 0; mf < 2; ++mf) {
#pragma unroll
    for (int df = 0; df < 4; ++df) o[mf][df] = vzero8();
#pragma unroll
    for (int e = 0; e < 8; ++e) { mrow[mf][e] = -3.0e38f; lrow[mf][e] = 0.f; }
  }

  const float scale = 0.03125f;              // 1/sqrt(E) = 1/32
  const int kb_end_w = (q0 + 95) / 64;       // this wave's causal bound
  const int KB = (q0b + 96 + 95) / 64;       // block bound (last wave's)

  stage(0, ldsK0, ldsV0);

  for (int kb = 0; kb < KB; ++kb) {
    const int kbase = kb * 64;
    const bf16_t* Kt = KtL[kb & 1];
    const bf16_t* Vt = VtL[kb & 1];

    // Prefetch next tile into the other buffer, wait for the current one.
    if (kb + 1 < KB) {
      stage(kbase + 64, (kb & 1) ? ldsK0 : ldsK1, (kb & 1) ? ldsV0 : ldsV1);
      asm volatile("s_wait_asynccnt 8" ::: "memory");
    } else {
      asm volatile("s_wait_asynccnt 0" ::: "memory");
    }
    __syncthreads();                          // all waves' chunks landed

    if (kb < kb_end_w) {
      const bool diag = (kbase + 63 > q0);

      // ---- scores S = Q K^T  (16 WMMAs) ----
      v8f s[2][4];
#pragma unroll
      for (int nf = 0; nf < 4; ++nf) {
        v16bf bk0 = load_b_frag(Kt + (nf * 16) * 64 + 0,  64);
        v16bf bk1 = load_b_frag(Kt + (nf * 16) * 64 + 32, 64);
#pragma unroll
        for (int mf = 0; mf < 2; ++mf) {
          v8f c = wmma_bf16(qf[mf][0], bk0, vzero8());
          s[mf][nf] = wmma_bf16(qf[mf][1], bk1, c);
        }
      }

      // ---- scale + causal mask + online softmax ----
#pragma unroll
      for (int mf = 0; mf < 2; ++mf) {
        float mnew[8];
#pragma unroll
        for (int e = 0; e < 8; ++e) {
          const int row = q0 + mf * 16 + half * 8 + e;
          float mx = mrow[mf][e];
#pragma unroll
          for (int nf = 0; nf < 4; ++nf) {
            float v = s[mf][nf][e] * scale;
            const int col = kbase + nf * 16 + nl;
            if (diag && col > row) v = -3.0e38f;
            s[mf][nf][e] = v;
            mx = fmaxf(mx, v);
          }
          mnew[e] = mx;
        }
#pragma unroll
        for (int e = 0; e < 8; ++e) {         // row max over 16-lane half
          float mx = mnew[e];
#pragma unroll
          for (int off = 1; off < 16; off <<= 1)
            mx = fmaxf(mx, __shfl_xor(mx, off, 32));
          mnew[e] = mx;
        }
        float corr[8];
#pragma unroll
        for (int e = 0; e < 8; ++e) {
          const float mn = mnew[e];
          corr[e] = __expf(mrow[mf][e] - mn);
          float rs = 0.f;
#pragma unroll
          for (int nf = 0; nf < 4; ++nf) {
            float p = __expf(s[mf][nf][e] - mn);
            s[mf][nf][e] = p;
            rs += p;
          }
#pragma unroll
          for (int off = 1; off < 16; off <<= 1) rs += __shfl_xor(rs, off, 32);
          lrow[mf][e] = lrow[mf][e] * corr[e] + rs;
          mrow[mf][e] = mn;
        }
#pragma unroll
        for (int df = 0; df < 4; ++df)
#pragma unroll
          for (int e = 0; e < 8; ++e) o[mf][df][e] *= corr[e];
      }

      // ---- P (C-layout) -> per-wave LDS [32 q][64 keys] bf16 ----
#pragma unroll
      for (int mf = 0; mf < 2; ++mf)
#pragma unroll
        for (int nf = 0; nf < 4; ++nf)
#pragma unroll
          for (int e = 0; e < 8; ++e)
            P[(mf * 16 + half * 8 + e) * 64 + nf * 16 + nl] = (bf16_t)s[mf][nf][e];
      asm volatile("s_wait_dscnt 0" ::: "memory");

      // ---- O += P x V  (16 WMMAs; A-frags from LDS, B-frags from V tile) ----
#pragma unroll
      for (int mf = 0; mf < 2; ++mf) {
        v16bf pf0 = load_a_frag(P + mf * 16 * 64 + 0,  64);
        v16bf pf1 = load_a_frag(P + mf * 16 * 64 + 32, 64);
#pragma unroll
        for (int df = 0; df < 4; ++df) {
          v16bf bv0 = load_b_frag(Vt + (df * 16) * 64 + 0,  64);
          v16bf bv1 = load_b_frag(Vt + (df * 16) * 64 + 32, 64);
          o[mf][df] = wmma_bf16(pf0, bv0, o[mf][df]);
          o[mf][df] = wmma_bf16(pf1, bv1, o[mf][df]);
        }
      }
    }
    __syncthreads();    // everyone done reading this buffer before restage
  }

  // ---- normalize and write [N,S,E] bf16 ----
#pragma unroll
  for (int mf = 0; mf < 2; ++mf)
#pragma unroll
    for (int e = 0; e < 8; ++e) {
      const float inv = 1.0f / lrow[mf][e];
      const int row = q0 + mf * 16 + half * 8 + e;      // s
#pragma unroll
      for (int df = 0; df < 4; ++df) {
        const int d = df * 16 + nl;
        Om[((size_t)nb * S + row) * 1024 + h * 64 + d] = (bf16_t)(o[mf][df][e] * inv);
      }
    }
}

// ---------------------------------------------------------------------------
// Host-side launcher
// ---------------------------------------------------------------------------
extern "C" void kernel_launch(void* const* d_in, const int* in_sizes, int n_in,
                              void* d_out, int out_size, void* d_ws, size_t ws_size,
                              hipStream_t stream) {
  (void)in_sizes; (void)n_in; (void)out_size; (void)ws_size;
  constexpr int Nb = 2, S = 2048, E = 1024;
  constexpr int M = Nb * S;                       // 4096 tokens
  const size_t XE = (size_t)M * E;
  const size_t WE = (size_t)E * E;

  const float* values = (const float*)d_in[0];
  const float* keys   = (const float*)d_in[1];
  const float* query  = (const float*)d_in[2];
  // d_in[3] = mask (unused: reference takes the causal path)
  const float* Wv = (const float*)d_in[4];  const float* bv = (const float*)d_in[5];
  const float* Wk = (const float*)d_in[6];  const float* bk = (const float*)d_in[7];
  const float* Wq = (const float*)d_in[8];  const float* bq = (const float*)d_in[9];
  const float* Wo = (const float*)d_in[10]; const float* bo = (const float*)d_in[11];

  char* w = (char*)d_ws;
  bf16_t* Xv  = (bf16_t*)w; w += XE * 2;
  bf16_t* Xk  = (bf16_t*)w; w += XE * 2;
  bf16_t* Xq  = (bf16_t*)w; w += XE * 2;
  bf16_t* Wvb = (bf16_t*)w; w += WE * 2;
  bf16_t* Wkb = (bf16_t*)w; w += WE * 2;
  bf16_t* Wqb = (bf16_t*)w; w += WE * 2;
  bf16_t* Wob = (bf16_t*)w; w += WE * 2;
  bf16_t* Qb  = (bf16_t*)w; w += XE * 2;          // [N,H,S,D]
  bf16_t* Kb  = (bf16_t*)w; w += XE * 2;          // [N,H,S,D]
  bf16_t* VTb = (bf16_t*)w; w += XE * 2;          // [N,H,D,S]
  bf16_t* Ab  = (bf16_t*)w; w += XE * 2;          // attention out [N,S,E]

  const int cgx = (int)((XE + 255) / 256);
  const int cgw = (int)((WE + 255) / 256);
  cvt_f32_bf16<<<cgx, 256, 0, stream>>>(values, Xv, (int)XE);
  cvt_f32_bf16<<<cgx, 256, 0, stream>>>(keys,   Xk, (int)XE);
  cvt_f32_bf16<<<cgx, 256, 0, stream>>>(query,  Xq, (int)XE);
  cvt_f32_bf16<<<cgw, 256, 0, stream>>>(Wv, Wvb, (int)WE);
  cvt_f32_bf16<<<cgw, 256, 0, stream>>>(Wk, Wkb, (int)WE);
  cvt_f32_bf16<<<cgw, 256, 0, stream>>>(Wq, Wqb, (int)WE);
  cvt_f32_bf16<<<cgw, 256, 0, stream>>>(Wo, Wob, (int)WE);

  dim3 ggrid(M / 256, E / 64);
  gemm_bf16_wmma<<<ggrid, 128, 0, stream>>>(Xq, Wqb, bq, Qb,  M, E, E, 1);
  gemm_bf16_wmma<<<ggrid, 128, 0, stream>>>(Xk, Wkb, bk, Kb,  M, E, E, 1);
  gemm_bf16_wmma<<<ggrid, 128, 0, stream>>>(Xv, Wvb, bv, VTb, M, E, E, 2);

  dim3 agrid(S / 128, Nb * 16);
  attn_flash_wmma<<<agrid, 128, 0, stream>>>(Qb, Kb, VTb, Ab);

  gemm_bf16_wmma<<<ggrid, 128, 0, stream>>>(Ab, Wob, bo, (float*)d_out, M, E, E, 0);
}
